// FeaturesAggregate_3100966388130
// MI455X (gfx1250) — compile-verified, matrix operand order checked by reading, not run on hardware
//
#include <hip/hip_runtime.h>

typedef __attribute__((ext_vector_type(16))) __bf16 v16bf;
typedef __attribute__((ext_vector_type(8)))  float  v8f;

#define EPS 1e-5f
#define ATT_SCALE 0.17677669529663687f  // (256/8)^-0.5
#define USE_ASYNC_LDS 1

// K-position of bf16 element pair j (VGPR j) for lane-half g, per CDNA5 ISA 7.12.2
__device__ __forceinline__ int kbase(int j, int g) {
    return 16 * (j >> 2) + 2 * (j & 3) + 8 * g;
}

#if USE_ASYNC_LDS
// GLOBAL_LOAD_ASYNC_TO_LDS_B128 (ISA 15.18.3 op 98): per-lane 16B global->LDS DMA,
// tracked by ASYNCcnt. vdst = LDS byte offset (addr[31:0] of flat shared address).
__device__ __forceinline__ void async_copy_b128(unsigned lds_off, const void* gaddr) {
    asm volatile("global_load_async_to_lds_b128 %0, %1, off"
                 :: "v"(lds_off), "v"(gaddr) : "memory");
}
__device__ __forceinline__ void wait_async() {
    asm volatile("s_wait_asynccnt 0x0" ::: "memory");
}
#endif

// ---------------------------------------------------------------------------
// Generic batched strided GEMM: C[b][m][n] = act( A[b][m][:] x B[:][n] + bias )
// ---------------------------------------------------------------------------
struct GemmP {
    const void* A; int a_bf16;
    long long As; int Ars, Acs;
    const float* B; int Bns, Bks;
    const float* bias;
    void* C; int c_bf16;
    long long Cs; int Crs, Ccs;
    const float* R; long long Rs; int Rrs, Rcs; int RdivB;
    int relu;
    const float* bn_g; const float* bn_b; const float* bn_m; const float* bn_v;
    const float* cb;   // conv bias folded into BN epilogue
    int M, N, K;
};

template <int ABF16>
__global__ void gemm_wmma(GemmP p) {
    __shared__ __align__(16) __bf16 At[64][40];  // [m][k], padded (80B row, 16B-aligned)
    __shared__ __align__(16) __bf16 Bt[64][40];  // [n][k], padded
    const int tid  = threadIdx.x;
    const int wave = tid >> 5, lane = tid & 31;
    const int g = lane >> 4, r16 = lane & 15;
    const int m0 = blockIdx.y * 64, n0 = blockIdx.x * 64;
    const int batch = blockIdx.z;
    const int wm = (wave >> 1) * 32, wn = (wave & 1) * 32;
    const long long aoff = (long long)batch * p.As;

    // fast-path eligibility (block-uniform)
    const int  row   = tid >> 1;             // 0..63
    const int  khalf = (tid & 1) * 16;       // 0 or 16
    const bool aFast = (p.Acs == 1) && (m0 + 64 <= p.M) &&
                       ((ABF16 ? (p.Ars & 7) : (p.Ars & 3)) == 0);
    const bool bFast = (p.Bks == 1) && (n0 + 64 <= p.N) && ((p.Bns & 3) == 0);

    v8f acc[2][2] = {};

    for (int k0 = 0; k0 < p.K; k0 += 32) {
        const bool fullK = (k0 + 32 <= p.K);

        // ---- stage A tile (64x32) ----
        if (aFast && fullK) {
            if (ABF16) {
                const __bf16* src = (const __bf16*)p.A + aoff +
                                    (long long)(m0 + row) * p.Ars + (k0 + khalf);
#if USE_ASYNC_LDS
                unsigned loff = (unsigned)(unsigned long long)&At[row][khalf];
                async_copy_b128(loff,      src);
                async_copy_b128(loff + 16, src + 8);
#else
                const uint4* s4 = (const uint4*)src;
                uint4 d0 = s4[0], d1 = s4[1];
                *(uint4*)&At[row][khalf]     = d0;
                *(uint4*)&At[row][khalf + 8] = d1;
#endif
            } else {
                const float* src = (const float*)p.A + aoff +
                                   (long long)(m0 + row) * p.Ars + (k0 + khalf);
                float4 f0 = ((const float4*)src)[0];
                float4 f1 = ((const float4*)src)[1];
                float4 f2 = ((const float4*)src)[2];
                float4 f3 = ((const float4*)src)[3];
                if (k0 + 32 < p.K) __builtin_prefetch((const char*)src + 128, 0, 1);
                __bf16* d = &At[row][khalf];
                d[0]=(__bf16)f0.x; d[1]=(__bf16)f0.y; d[2]=(__bf16)f0.z; d[3]=(__bf16)f0.w;
                d[4]=(__bf16)f1.x; d[5]=(__bf16)f1.y; d[6]=(__bf16)f1.z; d[7]=(__bf16)f1.w;
                d[8]=(__bf16)f2.x; d[9]=(__bf16)f2.y; d[10]=(__bf16)f2.z; d[11]=(__bf16)f2.w;
                d[12]=(__bf16)f3.x; d[13]=(__bf16)f3.y; d[14]=(__bf16)f3.z; d[15]=(__bf16)f3.w;
            }
        } else {
            // branchless guarded path: clamp index, select 0 if OOB
            for (int t = tid; t < 64 * 32; t += 128) {
                int mm = t >> 5, kk = t & 31;
                int gm = m0 + mm, gk = k0 + kk;
                bool ok = (gm < p.M) && (gk < p.K);
                int cm = gm < p.M ? gm : p.M - 1;
                int ck = gk < p.K ? gk : p.K - 1;
                long long ix = aoff + (long long)cm * p.Ars + (long long)ck * p.Acs;
                float v = ABF16 ? (float)((const __bf16*)p.A)[ix] : ((const float*)p.A)[ix];
                At[mm][kk] = (__bf16)(ok ? v : 0.f);
            }
        }
        // ---- stage B tile (32x64, [n][k]-major) ----
        if (bFast && fullK) {
            const float* src = p.B + (long long)(n0 + row) * p.Bns + (k0 + khalf);
            float4 f0 = ((const float4*)src)[0];
            float4 f1 = ((const float4*)src)[1];
            float4 f2 = ((const float4*)src)[2];
            float4 f3 = ((const float4*)src)[3];
            if (k0 + 32 < p.K) __builtin_prefetch((const char*)src + 128, 0, 1);
            __bf16* d = &Bt[row][khalf];
            d[0]=(__bf16)f0.x; d[1]=(__bf16)f0.y; d[2]=(__bf16)f0.z; d[3]=(__bf16)f0.w;
            d[4]=(__bf16)f1.x; d[5]=(__bf16)f1.y; d[6]=(__bf16)f1.z; d[7]=(__bf16)f1.w;
            d[8]=(__bf16)f2.x; d[9]=(__bf16)f2.y; d[10]=(__bf16)f2.z; d[11]=(__bf16)f2.w;
            d[12]=(__bf16)f3.x; d[13]=(__bf16)f3.y; d[14]=(__bf16)f3.z; d[15]=(__bf16)f3.w;
        } else {
            for (int t = tid; t < 64 * 32; t += 128) {
                int nn = t >> 5, kk = t & 31;
                int gn = n0 + nn, gk = k0 + kk;
                bool ok = (gn < p.N) && (gk < p.K);
                int cn = gn < p.N ? gn : p.N - 1;
                int ck = gk < p.K ? gk : p.K - 1;
                float v = p.B[(long long)cn * p.Bns + (long long)ck * p.Bks];
                Bt[nn][kk] = (__bf16)(ok ? v : 0.f);
            }
        }
#if USE_ASYNC_LDS
        if (ABF16 && aFast && fullK) wait_async();
#endif
        __syncthreads();
        #pragma unroll
        for (int ti = 0; ti < 2; ++ti) {
            v16bf a;
            #pragma unroll
            for (int j = 0; j < 8; ++j) {
                int kb = kbase(j, g);
                a[2 * j]     = At[wm + ti * 16 + r16][kb];
                a[2 * j + 1] = At[wm + ti * 16 + r16][kb + 1];
            }
            #pragma unroll
            for (int tj = 0; tj < 2; ++tj) {
                v16bf b;
                #pragma unroll
                for (int j = 0; j < 8; ++j) {
                    int kb = kbase(j, g);
                    b[2 * j]     = Bt[wn + tj * 16 + r16][kb];
                    b[2 * j + 1] = Bt[wn + tj * 16 + r16][kb + 1];
                }
                acc[ti][tj] = __builtin_amdgcn_wmma_f32_16x16x32_bf16(
                    false, a, false, b, (short)0, acc[ti][tj], false, false);
            }
        }
        __syncthreads();
    }

    // epilogue: D element e -> M = e + 8*(lane>>4), N = lane&15 (ISA 7.12.2)
    #pragma unroll
    for (int ti = 0; ti < 2; ++ti)
    #pragma unroll
    for (int tj = 0; tj < 2; ++tj)
    #pragma unroll
    for (int e = 0; e < 8; ++e) {
        int m = m0 + wm + ti * 16 + 8 * g + e;
        int n = n0 + wn + tj * 16 + r16;
        if (m < p.M && n < p.N) {
            float v = acc[ti][tj][e];
            if (p.bias) v += p.bias[n];
            if (p.bn_g) {
                float s = p.bn_g[n] * rsqrtf(p.bn_v[n] + EPS);
                v = (v + (p.cb ? p.cb[n] : 0.f) - p.bn_m[n]) * s + p.bn_b[n];
            }
            if (p.relu) v = fmaxf(v, 0.f);
            if (p.R) {
                long long ro = (long long)(batch / p.RdivB) * p.Rs +
                               (long long)m * p.Rrs + (long long)n * p.Rcs;
                v += p.R[ro];
            }
            long long co = (long long)batch * p.Cs + (long long)m * p.Crs + (long long)n * p.Ccs;
            if (p.c_bf16) ((__bf16*)p.C)[co] = (__bf16)v;
            else          ((float*)p.C)[co]  = v;
        }
    }
}

// ---------------------------------------------------------------------------
// Multi-head attention for one (batch, ref, head, 32-row query tile).
// ---------------------------------------------------------------------------
__global__ void attn_kernel(const __bf16* __restrict__ q, const __bf16* __restrict__ k,
                            const __bf16* __restrict__ vT, const int* __restrict__ idx,
                            __bf16* __restrict__ o, int idx_off) {
    __shared__ float  S[32][256];
    __shared__ __bf16 P[32][256];
    __shared__ float  mred[64], sred[64];
    const int tid = threadIdx.x;
    const int w = tid >> 5, lane = tid & 31;
    const int g = lane >> 4, r16 = lane & 15;
    const int n0 = blockIdx.x * 32;
    const int h  = blockIdx.y;
    const int b  = blockIdx.z >> 2, rr = blockIdx.z & 3;
    const int bp = idx[idx_off + b * 4 + rr];

    // Q fragments for full K=256 held in registers
    const long long qrow = ((long long)b * 256 + (n0 + w * 16 + r16)) * 2048 + h * 256;
    v16bf af[8];
    #pragma unroll
    for (int s = 0; s < 8; ++s)
        #pragma unroll
        for (int j = 0; j < 8; ++j) {
            int kb = s * 32 + kbase(j, g);
            af[s][2 * j]     = q[qrow + kb];
            af[s][2 * j + 1] = q[qrow + kb + 1];
        }

    // S = (q . k^T) * scale
    const __bf16* kb_ = k + (long long)bp * 524288 + h * 256;
    for (int mt = 0; mt < 16; ++mt) {
        v8f acc = {};
        const long long kcol = (long long)(mt * 16 + r16) * 2048;
        #pragma unroll
        for (int s = 0; s < 8; ++s) {
            v16bf bf;
            #pragma unroll
            for (int j = 0; j < 8; ++j) {
                int kk = s * 32 + kbase(j, g);
                bf[2 * j]     = kb_[kcol + kk];
                bf[2 * j + 1] = kb_[kcol + kk + 1];
            }
            acc = __builtin_amdgcn_wmma_f32_16x16x32_bf16(false, af[s], false, bf,
                                                          (short)0, acc, false, false);
        }
        #pragma unroll
        for (int e = 0; e < 8; ++e)
            S[w * 16 + 8 * g + e][mt * 16 + r16] = acc[e] * ATT_SCALE;
    }
    __syncthreads();

    // row softmax over m (keys); 2 threads per row, LDS combine (no divergence)
    {
        const int row = tid >> 1, half = (tid & 1) * 128;
        float mx = -1e30f;
        for (int m = 0; m < 128; ++m) mx = fmaxf(mx, S[row][half + m]);
        mred[tid] = mx; __syncthreads();
        float rmx = fmaxf(mred[row * 2], mred[row * 2 + 1]);
        float sum = 0.f;
        for (int m = 0; m < 128; ++m) {
            float e = __expf(S[row][half + m] - rmx);
            S[row][half + m] = e; sum += e;
        }
        sred[tid] = sum; __syncthreads();
        float inv = 1.f / (sred[row * 2] + sred[row * 2 + 1]);
        for (int m = 0; m < 128; ++m)
            P[row][half + m] = (__bf16)(S[row][half + m] * inv);
    }
    __syncthreads();

    // O = P . V  (K = m = 256)
    v16bf pf[8];
    #pragma unroll
    for (int s = 0; s < 8; ++s)
        #pragma unroll
        for (int j = 0; j < 8; ++j) {
            int kk = s * 32 + kbase(j, g);
            pf[s][2 * j]     = P[w * 16 + r16][kk];
            pf[s][2 * j + 1] = P[w * 16 + r16][kk + 1];
        }
    const __bf16* vb = vT + (long long)bp * 524288 + (long long)h * 256 * 256;
    for (int dt = 0; dt < 16; ++dt) {
        v8f acc = {};
        const long long dcol = (long long)(dt * 16 + r16) * 256;
        #pragma unroll
        for (int s = 0; s < 8; ++s) {
            v16bf bf;
            #pragma unroll
            for (int j = 0; j < 8; ++j) {
                int kk = s * 32 + kbase(j, g);
                bf[2 * j]     = vb[dcol + kk];
                bf[2 * j + 1] = vb[dcol + kk + 1];
            }
            acc = __builtin_amdgcn_wmma_f32_16x16x32_bf16(false, pf[s], false, bf,
                                                          (short)0, acc, false, false);
        }
        #pragma unroll
        for (int e = 0; e < 8; ++e) {
            int n = n0 + w * 16 + 8 * g + e;
            o[(((long long)b * 4 + rr) * 256 + n) * 2048 + h * 256 + dt * 16 + r16] = (__bf16)acc[e];
        }
    }
}

// LayerNorm over last dim (256); one row per block of 256 threads
__global__ void ln_kernel(const float* __restrict__ y, const float* __restrict__ g,
                          const float* __restrict__ b, __bf16* __restrict__ out) {
    __shared__ float red[256];
    const int row = blockIdx.x, t = threadIdx.x;
    float v = y[(long long)row * 256 + t];
    red[t] = v; __syncthreads();
    for (int s = 128; s > 0; s >>= 1) { if (t < s) red[t] += red[t + s]; __syncthreads(); }
    float mu = red[0] * (1.f / 256.f); __syncthreads();
    float d = v - mu;
    red[t] = d * d; __syncthreads();
    for (int s = 128; s > 0; s >>= 1) { if (t < s) red[t] += red[t + s]; __syncthreads(); }
    float var = red[0] * (1.f / 256.f);
    out[(long long)row * 256 + t] = (__bf16)(d * rsqrtf(var + EPS) * g[t] + b[t]);
}

// 3x3 conv pad 1, 256->256, + BN + ReLU (direct; only ~5 GFLOP total)
__global__ void conv3x3_kernel(const float* __restrict__ in, const float* __restrict__ w,
                               const float* __restrict__ cb, const float* __restrict__ bg,
                               const float* __restrict__ bb, const float* __restrict__ bm,
                               const float* __restrict__ bv, float* __restrict__ out, int s) {
    const int o = blockIdx.y, b = blockIdx.z;
    const int p = blockIdx.x * blockDim.x + threadIdx.x;
    const int ss = s * s;
    if (p >= ss) return;
    const int px = p / s, py = p % s;
    float acc = cb[o];
    const float* wo = w + (long long)o * 256 * 9;
    const float* ib = in + (long long)b * 256 * ss;
    for (int c = 0; c < 256; ++c) {
        const float* ic = ib + c * ss;
        const float* wc = wo + c * 9;
        #pragma unroll
        for (int dx = -1; dx <= 1; ++dx) {
            int xx = px + dx; if ((unsigned)xx >= (unsigned)s) continue;
            #pragma unroll
            for (int dy = -1; dy <= 1; ++dy) {
                int yy = py + dy; if ((unsigned)yy >= (unsigned)s) continue;
                acc += ic[xx * s + yy] * wc[(dx + 1) * 3 + (dy + 1)];
            }
        }
    }
    float sc = bg[o] * rsqrtf(bv[o] + EPS);
    float r = (acc - bm[o]) * sc + bb[o];
    out[((long long)b * 256 + o) * ss + p] = fmaxf(r, 0.f);
}

// ---------------------------------------------------------------------------
static void launch_gemm(hipStream_t st, const GemmP& p, int batch) {
    dim3 grid((p.N + 63) / 64, (p.M + 63) / 64, batch);
    if (p.a_bf16)
        hipLaunchKernelGGL((gemm_wmma<1>), grid, dim3(128), 0, st, p);
    else
        hipLaunchKernelGGL((gemm_wmma<0>), grid, dim3(128), 0, st, p);
}

extern "C" void kernel_launch(void* const* d_in, const int* in_sizes, int n_in,
                              void* d_out, int out_size, void* d_ws, size_t ws_size,
                              hipStream_t stream) {
    (void)in_sizes; (void)n_in; (void)out_size; (void)ws_size;
    const float* x1   = (const float*)d_in[0];
    const float* x2   = (const float*)d_in[1];
    const float* x3   = (const float*)d_in[2];
    const int*   ridx = (const int*)d_in[3];
    const float* wq = (const float*)d_in[4],  *bq = (const float*)d_in[5];
    const float* wk = (const float*)d_in[6],  *bk = (const float*)d_in[7];
    const float* wv = (const float*)d_in[8],  *bv = (const float*)d_in[9];
    const float* wp = (const float*)d_in[10], *bp = (const float*)d_in[11];
    const float* ln_g = (const float*)d_in[12], *ln_b = (const float*)d_in[13];
    const float* x1rw = (const float*)d_in[14], *x1rb = (const float*)d_in[15];
    const float* c1w = (const float*)d_in[16], *c1b = (const float*)d_in[17];
    const float* c2w = (const float*)d_in[18], *c2b = (const float*)d_in[19];
    const float* c3w = (const float*)d_in[20], *c3b = (const float*)d_in[21];
    const float* bn1[4] = {(const float*)d_in[22], (const float*)d_in[23], (const float*)d_in[24], (const float*)d_in[25]};
    const float* bn2[4] = {(const float*)d_in[26], (const float*)d_in[27], (const float*)d_in[28], (const float*)d_in[29]};
    const float* bn3[4] = {(const float*)d_in[30], (const float*)d_in[31], (const float*)d_in[32], (const float*)d_in[33]};
    const float* align_w[3], *align_b[3], *rec_w[3], *rec_b[3];
    for (int i = 0; i < 3; ++i) {
        align_w[i] = (const float*)d_in[34 + 4 * i];
        align_b[i] = (const float*)d_in[35 + 4 * i];
        rec_w[i]   = (const float*)d_in[36 + 4 * i];
        rec_b[i]   = (const float*)d_in[37 + 4 * i];
    }

    size_t off = 0;
    auto wsalloc = [&](size_t bytes) {
        void* p = (char*)d_ws + off;
        off += (bytes + 255) & ~(size_t)255;
        return p;
    };
    float* x1r  = (float*)wsalloc(32ull * 256 * 100 * 4);
    float* xa   = (float*)wsalloc(32ull * 256 * 256 * 4);
    __bf16* qb  = (__bf16*)wsalloc(32ull * 256 * 2048 * 2);
    __bf16* kb  = (__bf16*)wsalloc(32ull * 256 * 2048 * 2);
    __bf16* vtb = (__bf16*)wsalloc(32ull * 2048 * 256 * 2);
    __bf16* ob  = (__bf16*)wsalloc(32ull * 4 * 256 * 2048 * 2);
    float* yb   = (float*)wsalloc(32ull * 4 * 256 * 256 * 4);
    __bf16* yln = (__bf16*)wsalloc(32ull * 4 * 256 * 256 * 2);
    float* fb   = (float*)wsalloc(32ull * 4 * 256 * 100 * 4);
    float* o1b  = (float*)wsalloc(32ull * 256 * 100 * 4);
    float* o2b  = (float*)wsalloc(32ull * 256 * 100 * 4);

    const int F[3] = {100, 25, 9};
    const int SD[3] = {10, 5, 3};
    float* outp = (float*)d_out;
    float* outs[3] = {outp, outp + 32 * 256 * 100, outp + 32 * 256 * 125};

    GemmP z = {}; z.RdivB = 1;

    // stage 0: x1 reduce (1x1 conv 512->256)
    {
        GemmP p = z;
        p.A = x1; p.As = 512ll * 100; p.Ars = 1; p.Acs = 100;
        p.B = x1rw; p.Bns = 512; p.Bks = 1; p.bias = x1rb;
        p.C = x1r; p.Cs = 256ll * 100; p.Crs = 1; p.Ccs = 100;
        p.M = 100; p.N = 256; p.K = 512;
        launch_gemm(stream, p, 32);
    }

    const float* xs[3] = {x1r, x2, x3};

    for (int i = 0; i < 3; ++i) {
        const int f = F[i], s = SD[i];

        // 1) align
        { GemmP p = z;
          p.A = xs[i]; p.As = 256ll * f; p.Ars = f; p.Acs = 1;
          p.B = align_w[i]; p.Bns = f; p.Bks = 1; p.bias = align_b[i];
          p.C = xa; p.Cs = 65536; p.Crs = 256; p.Ccs = 1;
          p.M = 256; p.N = 256; p.K = f;
          launch_gemm(stream, p, 32); }

        // 2-4) q/k/v projections; v stored transposed
        { GemmP p = z;
          p.A = xa; p.As = 65536; p.Ars = 256; p.Acs = 1;
          p.B = wq; p.Bns = 256; p.Bks = 1; p.bias = bq;
          p.C = qb; p.c_bf16 = 1; p.Cs = 524288; p.Crs = 2048; p.Ccs = 1;
          p.M = 256; p.N = 2048; p.K = 256;
          launch_gemm(stream, p, 32);
          p.B = wk; p.bias = bk; p.C = kb;
          launch_gemm(stream, p, 32);
          p.B = wv; p.bias = bv; p.C = vtb; p.Crs = 1; p.Ccs = 256;
          launch_gemm(stream, p, 32); }

        // 5) attention (gathers refs via rand_idx)
        hipLaunchKernelGGL(attn_kernel, dim3(8, 8, 128), dim3(64), 0, stream,
                           qb, kb, vtb, ridx, ob, i * 128);

        // 6) y = relu(o . Wp^T + bp) + xa
        { GemmP p = z;
          p.A = ob; p.a_bf16 = 1; p.As = 524288; p.Ars = 2048; p.Acs = 1;
          p.B = wp; p.Bns = 2048; p.Bks = 1; p.bias = bp; p.relu = 1;
          p.R = xa; p.Rs = 65536; p.Rrs = 256; p.Rcs = 1; p.RdivB = 4;
          p.C = yb; p.Cs = 65536; p.Crs = 256; p.Ccs = 1;
          p.M = 256; p.N = 256; p.K = 2048;
          launch_gemm(stream, p, 128); }

        // 7) LayerNorm -> bf16
        hipLaunchKernelGGL(ln_kernel, dim3(32 * 4 * 256), dim3(256), 0, stream,
                           yb, ln_g, ln_b, yln);

        // 8) out = x + yln . Wr^T + br  -> fb (B,4*256,f)
        { GemmP p = z;
          p.A = yln; p.a_bf16 = 1; p.As = 65536; p.Ars = 256; p.Acs = 1;
          p.B = rec_w[i]; p.Bns = 256; p.Bks = 1; p.bias = rec_b[i];
          p.R = xs[i]; p.Rs = 256ll * f; p.Rrs = f; p.Rcs = 1; p.RdivB = 4;
          p.C = fb; p.Cs = 1024ll * f; p.Crs = f; p.Ccs = 1;
          p.M = 256; p.N = f; p.K = 256;
          launch_gemm(stream, p, 128); }

        // 9) conv1 1x1 (1024->256) + BN1 + ReLU
        { GemmP p = z;
          p.A = fb; p.As = 1024ll * f; p.Ars = 1; p.Acs = f;
          p.B = c1w; p.Bns = 1024; p.Bks = 1;
          p.bn_g = bn1[0]; p.bn_b = bn1[1]; p.bn_m = bn1[2]; p.bn_v = bn1[3];
          p.cb = c1b; p.relu = 1;
          p.C = o1b; p.Cs = 256ll * f; p.Crs = 1; p.Ccs = f;
          p.M = f; p.N = 256; p.K = 1024;
          launch_gemm(stream, p, 32); }

        // 10) conv2 3x3 pad1 + BN2 + ReLU
        hipLaunchKernelGGL(conv3x3_kernel, dim3((f + 63) / 64, 256, 32), dim3(64), 0,
                           stream, o1b, c2w, c2b, bn2[0], bn2[1], bn2[2], bn2[3], o2b, s);

        // 11) conv3 1x1 (256->256) + BN3 + ReLU -> final output
        { GemmP p = z;
          p.A = o2b; p.As = 256ll * f; p.Ars = 1; p.Acs = f;
          p.B = c3w; p.Bns = 256; p.Bks = 1;
          p.bn_g = bn3[0]; p.bn_b = bn3[1]; p.bn_m = bn3[2]; p.bn_v = bn3[3];
          p.cb = c3b; p.relu = 1;
          p.C = outs[i]; p.Cs = 256ll * f; p.Crs = 1; p.Ccs = f;
          p.M = f; p.N = 256; p.K = 256;
          launch_gemm(stream, p, 32); }
    }
}
